// SelfAttention_44040594653423
// MI455X (gfx1250) — matrix operand order repaired and checked by reading, not compiled, not run on hardware
//
#include <hip/hip_runtime.h>
#include <hip/hip_bf16.h>

typedef __attribute__((ext_vector_type(16))) __bf16 v16bf;
typedef __attribute__((ext_vector_type(8)))  float  v8f;
typedef __attribute__((ext_vector_type(4)))  unsigned int v4u;
typedef __attribute__((ext_vector_type(8)))  int v8i;
typedef __attribute__((ext_vector_type(4)))  int v4i;

union UAP { uint4 q2[2]; v16bf b; unsigned short h[16]; };

#define NSP 4096   // spatial positions D*H*W
#define CCH 256    // channels C
#define CQ  32     // reduced channels Cq
#define NB  4      // batch

__device__ __forceinline__ unsigned short f2bf(float f) {
  unsigned int u = __float_as_uint(f);
  u += 0x7FFFu + ((u >> 16) & 1u);      // round-to-nearest-even
  return (unsigned short)(u >> 16);
}

// ---- Tensor Data Mover: 2D tile load (data_size = 4B), D# per ISA 8.3/8.4 ----
__device__ __forceinline__ void tdm_load_2d(unsigned lds_off, const void* gaddr,
                                            unsigned tensor_d0, unsigned tensor_d1,
                                            unsigned tile_d0, unsigned tile_d1,
                                            unsigned long stride0) {  // all in 4B units
  const unsigned long ga = (unsigned long)gaddr;
  v4u g0;
  g0[0] = 1u;                                            // count=1, user descriptor
  g0[1] = lds_off;                                       // LDS byte address
  g0[2] = (unsigned)(ga & 0xFFFFFFFFu);                  // global_addr[31:0]
  g0[3] = (unsigned)((ga >> 32) & 0x01FFFFFFu) | 0x80000000u; // addr[56:32], type=2
  v8i g1;
  g1[0] = (int)(2u << 16);                               // wg_mask=0, data_size=4B
  g1[1] = (int)((tensor_d0 & 0xFFFFu) << 16);            // barrier_addr=0, dim0[15:0]
  g1[2] = (int)((tensor_d0 >> 16) | ((tensor_d1 & 0xFFFFu) << 16));
  g1[3] = (int)((tensor_d1 >> 16) | (tile_d0 << 16));    // dim1[31:16], tile_dim0
  g1[4] = (int)(tile_d1 & 0xFFFFu);                      // tile_dim1, tile_dim2=0
  g1[5] = (int)(stride0 & 0xFFFFFFFFu);                  // dim0_stride[31:0]
  g1[6] = (int)((stride0 >> 32) & 0xFFFFu);              // dim0_stride[47:32]
  g1[7] = 0;
  const v4i z4 = {0, 0, 0, 0};
  const v8i z8 = {0, 0, 0, 0, 0, 0, 0, 0};
  __builtin_amdgcn_tensor_load_to_lds(g0, g1, z4, z4, z8, 0);
}

// ---------------- fragment staging (one wave packs one 1KB fragment tile) ----------------
__global__ void stage_afrag(const float* __restrict__ src, unsigned short* __restrict__ dst,
                            int KT, int ldr, int ldk, long bsSrc, long bsDst) {
  const int lane = threadIdx.x & 31;
  const long tile = (long)blockIdx.x * 4 + (threadIdx.x >> 5);
  const int mt = (int)(tile / KT), kt = (int)(tile % KT);
  const float* s = src + (long)blockIdx.y * bsSrc;
  const int row = mt * 16 + (lane & 15);
  const int kb = (lane < 16) ? 0 : 8;
  UAP fp;
#pragma unroll
  for (int e = 0; e < 16; ++e) {
    const int ka = kb + e + ((e >= 8) ? 8 : 0);        // ISA 16-bit A layout
    fp.h[e] = f2bf(s[(long)row * ldr + (long)(kt * 32 + ka) * ldk]);
  }
  uint4* dp = (uint4*)(dst + (long)blockIdx.y * bsDst + tile * 512 + lane * 16);
  dp[0] = fp.q2[0]; dp[1] = fp.q2[1];
}

__global__ void stage_bfrag(const float* __restrict__ src, unsigned short* __restrict__ dst,
                            int KT, int ldk, int ldc, long bsSrc, long bsDst) {
  const int lane = threadIdx.x & 31;
  const long tile = (long)blockIdx.x * 4 + (threadIdx.x >> 5);
  const int nt = (int)(tile / KT), kt = (int)(tile % KT);
  const float* s = src + (long)blockIdx.y * bsSrc;
  const int col = nt * 16 + (lane & 15);
  const int kb = (lane < 16) ? 0 : 8;
  UAP fp;
#pragma unroll
  for (int e = 0; e < 16; ++e)                          // ISA 16-bit B layout
    fp.h[e] = f2bf(s[(long)(kt * 32 + kb + e + ((e >= 8) ? 8 : 0)) * ldk + (long)col * ldc]);
  uint4* dp = (uint4*)(dst + (long)blockIdx.y * bsDst + tile * 512 + lane * 16);
  dp[0] = fp.q2[0]; dp[1] = fp.q2[1];
}

// ---------------- fragment GEMM: Y[b][m][n] = sum_k A[m][k]*B[b][k][n] ----------------
__global__ void wmma_gemm_frag(const unsigned short* __restrict__ A,
                               const unsigned short* __restrict__ Bf,
                               float* __restrict__ Y, int KT, int ldY,
                               long bsB, long bsY) {
  const int lane = threadIdx.x & 31;
  const int nt = blockIdx.x * 4 + (threadIdx.x >> 5);
  const int mt = blockIdx.y;
  const unsigned short* abase = A + (long)mt * KT * 512 + lane * 16;
  const unsigned short* bbase = Bf + (long)blockIdx.z * bsB + (long)nt * KT * 512 + lane * 16;
  float* Yb = Y + (long)blockIdx.z * bsY;

  v8f acc = {};
  for (int kt = 0; kt < KT; ++kt) {
    UAP a, bm;
    const uint4* ap = (const uint4*)(abase + kt * 512);
    const uint4* bp = (const uint4*)(bbase + kt * 512);
    a.q2[0] = ap[0];  a.q2[1] = ap[1];
    bm.q2[0] = bp[0]; bm.q2[1] = bp[1];
    if (kt + 1 < KT) __builtin_prefetch(bbase + (kt + 1) * 512, 0, 1);
    acc = __builtin_amdgcn_wmma_f32_16x16x32_bf16(false, a.b, false, bm.b,
                                                  (short)0, acc, false, false);
  }
  const int col = lane & 15;
#pragma unroll
  for (int j = 0; j < 8; ++j) {
    const int r = (lane < 16) ? j : (j + 8);            // C-matrix row layout
    Yb[(long)(mt * 16 + r) * ldY + nt * 16 + col] = acc[j];
  }
}

// ---------------- flash attention: TDM double-buffered K/V fragments ----------------
// per wave: 16 queries x 128 channels; d_k=32, d_v=256
__global__ void attn_flash(const unsigned short* __restrict__ Qf,
                           const unsigned short* __restrict__ Kf,
                           const unsigned short* __restrict__ Vf,
                           float* __restrict__ Pre) {
  // LDS: [0,1KB) P staging | [1KB,5KB) K double buffer | [5KB,21KB) V double buffer
  __shared__ __align__(16) unsigned char lds[1024 + 2 * 2048 + 2 * 8192];
  unsigned short* pbuf = (unsigned short*)lds;
  const unsigned lds_base = (unsigned)(unsigned long)(&lds[0]); // flat addr[31:0] = LDS offset

  const int lane = threadIdx.x;
  const int col = lane & 15;
  const int nt = blockIdx.y;
  const long b = blockIdx.z;
  const float scale = 0.17677669529663687f;  // 1/sqrt(32)

  const unsigned short* kglob = Kf + b * (256L * 512);
  const unsigned short* vglob = Vf + b * (2048L * 512) + (long)blockIdx.x * 8 * 128 * 512;

  UAP aq;
  {
    const uint4* qp = (const uint4*)(Qf + b * (256L * 512) + (long)nt * 512 + lane * 16);
    aq.q2[0] = qp[0]; aq.q2[1] = qp[1];
  }

  float mrow[8], lrow[8];
  v8f o[8];
#pragma unroll
  for (int j = 0; j < 8; ++j) { mrow[j] = -__builtin_inff(); lrow[j] = 0.f; v8f z = {}; o[j] = z; }

  // preload chunk 0: K chunk = 2KB contiguous (1 row x 512 dw), V chunk = 8 rows x 256 dw,
  // row stride 128 tiles * 256 dw (all 4-byte units)
  tdm_load_2d(lds_base + 1024, kglob, 512, 1, 512, 1, 512);
  tdm_load_2d(lds_base + 5120, vglob, 256, 8, 256, 8, 32768);

  int cur = 0;
  for (int mk = 0; mk < 128; ++mk) {
    const int nxt = cur ^ 1;
    if (mk + 1 < 128) {   // issue next chunk, then wait for current (TDM is in-order)
      tdm_load_2d(lds_base + 1024 + nxt * 2048, kglob + (long)(mk + 1) * 1024,
                  512, 1, 512, 1, 512);
      tdm_load_2d(lds_base + 5120 + nxt * 8192, vglob + (long)(mk + 1) * 512,
                  256, 8, 256, 8, 32768);
      __builtin_amdgcn_s_wait_tensorcnt(2);
    } else {
      __builtin_amdgcn_s_wait_tensorcnt(0);
    }
    const unsigned char* kb_lds = lds + 1024 + cur * 2048 + lane * 32;
    const unsigned char* vb_lds = lds + 5120 + cur * 8192 + lane * 32;

    // logits S[16q x 32m]: two WMMAs from K fragments in LDS
    v8f s[2];
#pragma unroll
    for (int t = 0; t < 2; ++t) {
      UAP bk;
      const uint4* kp = (const uint4*)(kb_lds + t * 1024);
      bk.q2[0] = kp[0]; bk.q2[1] = kp[1];
      v8f z = {};
      s[t] = __builtin_amdgcn_wmma_f32_16x16x32_bf16(false, aq.b, false, bk.b,
                                                     (short)0, z, false, false);
    }
    // online softmax: row r's 16 cols live in one 16-lane half at VGPR j
    float corr[8];
#pragma unroll
    for (int j = 0; j < 8; ++j) {
      float s0 = s[0][j] * scale, s1 = s[1][j] * scale;
      float mx = fmaxf(s0, s1);
      mx = fmaxf(mx, __shfl_xor(mx, 1));
      mx = fmaxf(mx, __shfl_xor(mx, 2));
      mx = fmaxf(mx, __shfl_xor(mx, 4));
      mx = fmaxf(mx, __shfl_xor(mx, 8));
      const float mn = fmaxf(mrow[j], mx);
      corr[j] = __expf(mrow[j] - mn);
      mrow[j] = mn;
      const float p0 = __expf(s0 - mn);
      const float p1 = __expf(s1 - mn);
      s[0][j] = p0; s[1][j] = p1;
      float rs = p0 + p1;
      rs += __shfl_xor(rs, 1);
      rs += __shfl_xor(rs, 2);
      rs += __shfl_xor(rs, 4);
      rs += __shfl_xor(rs, 8);
      lrow[j] = lrow[j] * corr[j] + rs;
    }
#pragma unroll
    for (int ct = 0; ct < 8; ++ct)
#pragma unroll
      for (int j = 0; j < 8; ++j) o[ct][j] *= corr[j];

    // stage P (C layout) -> LDS bf16 -> reload in A layout (same-wave DS in-order)
#pragma unroll
    for (int j = 0; j < 8; ++j) {
      const int r = (lane < 16) ? j : (j + 8);
      pbuf[r * 32 + col]      = f2bf(s[0][j]);
      pbuf[r * 32 + 16 + col] = f2bf(s[1][j]);
    }
    UAP pa;
    {
      const int off0 = (lane < 16) ? 0 : 8;
      pa.q2[0] = *(const uint4*)&pbuf[col * 32 + off0];
      pa.q2[1] = *(const uint4*)&pbuf[col * 32 + off0 + 16];
    }
    // O[16q x 128c] += P * V^T : 8 WMMAs from V fragments in LDS
#pragma unroll
    for (int ct = 0; ct < 8; ++ct) {
      UAP bv;
      const uint4* vp = (const uint4*)(vb_lds + ct * 1024);
      bv.q2[0] = vp[0]; bv.q2[1] = vp[1];
      o[ct] = __builtin_amdgcn_wmma_f32_16x16x32_bf16(false, pa.b, false, bv.b,
                                                      (short)0, o[ct], false, false);
    }
    cur = nxt;
  }
  // normalize and store Pre[b][n][c] (coalesced across lanes)
  float* preb = Pre + b * ((long)NSP * CCH);
  const int c0 = blockIdx.x * 128;
#pragma unroll
  for (int j = 0; j < 8; ++j) {
    const float rl = __builtin_amdgcn_rcpf(lrow[j]);
    const int r = (lane < 16) ? j : (j + 8);
#pragma unroll
    for (int ct = 0; ct < 8; ++ct)
      preb[(long)(nt * 16 + r) * CCH + c0 + ct * 16 + col] = o[ct][j] * rl;
  }
}

// ---------------- InstanceNorm3d (biased var) + residual ----------------
__global__ void instnorm_add(const float* __restrict__ Y2, const float* __restrict__ X,
                             float* __restrict__ Out) {
  __shared__ float rs[256], rq[256];
  const int t = threadIdx.x;
  const long base = ((long)blockIdx.y * CCH + blockIdx.x) * NSP;
  const float* y = Y2 + base;
  const float* x = X + base;
  float* out = Out + base;
  float s = 0.f, q = 0.f;
  for (int n = t; n < NSP; n += 256) { const float v = y[n]; s += v; q += v * v; }
  rs[t] = s; rq[t] = q;
  __syncthreads();
  for (int off = 128; off > 0; off >>= 1) {
    if (t < off) { rs[t] += rs[t + off]; rq[t] += rq[t + off]; }
    __syncthreads();
  }
  const float mean = rs[0] * (1.f / NSP);
  const float var = rq[0] * (1.f / NSP) - mean * mean;
  const float rstd = rsqrtf(var + 1e-5f);
  for (int n = t; n < NSP; n += 256) out[n] = (y[n] - mean) * rstd + x[n];
}

extern "C" void kernel_launch(void* const* d_in, const int* in_sizes, int n_in,
                              void* d_out, int out_size, void* d_ws, size_t ws_size,
                              hipStream_t stream) {
  const float* x  = (const float*)d_in[0];
  const float* Wq = (const float*)d_in[1];
  const float* Wk = (const float*)d_in[2];
  const float* Wv = (const float*)d_in[3];
  const float* Wp = (const float*)d_in[4];

  float* outq = (float*)d_out;                       // [B, Cq, N]
  float* outk = outq + (long)NB * CQ * NSP;
  float* outv = outk + (long)NB * CQ * NSP;          // [B, C, N]
  float* outa = outv + (long)NB * CCH * NSP;         // [B, C, N]

  // workspace layout
  float* pre = (float*)d_ws;                                    // [B,N,C] f32
  float* y2  = pre + (long)NB * NSP * CCH;                      // [B,C,N] f32
  unsigned short* xF   = (unsigned short*)(y2 + (long)NB * NSP * CCH);
  unsigned short* qF   = xF  + (long)NB * 2048 * 512;           // x  B-frags (8 KT x 256 NT)
  unsigned short* kF   = qF  + (long)NB * 256 * 512;            // q  A-frags (256 tiles)
  unsigned short* vF   = kF  + (long)NB * 256 * 512;            // k  B-frags (256 tiles)
  unsigned short* preF = vF  + (long)NB * 2048 * 512;           // v  B-frags (16x128 tiles)
  unsigned short* WqF  = preF + (long)NB * 2048 * 512;          // pre B-frags
  unsigned short* WkF  = WqF + 16 * 512;
  unsigned short* WvF  = WkF + 16 * 512;
  unsigned short* WpF  = WvF + 128 * 512;

  const dim3 blk(128);
  // stage weights (A-operand, row-major [M,K]: ldr=K, ldk=1)
  stage_afrag<<<dim3(4, 1),  blk, 0, stream>>>(Wq, WqF, 8, CCH, 1, 0, 0);
  stage_afrag<<<dim3(4, 1),  blk, 0, stream>>>(Wk, WkF, 8, CCH, 1, 0, 0);
  stage_afrag<<<dim3(32, 1), blk, 0, stream>>>(Wv, WvF, 8, CCH, 1, 0, 0);
  stage_afrag<<<dim3(32, 1), blk, 0, stream>>>(Wp, WpF, 8, CCH, 1, 0, 0);
  // stage x as B-operand: B[k=c][col=n] = x[c*N+n]
  stage_bfrag<<<dim3(512, NB), blk, 0, stream>>>(x, xF, 8, NSP, 1,
                                                 (long)CCH * NSP, 2048L * 512);
  // q/k/v projections
  wmma_gemm_frag<<<dim3(64, CQ / 16,  NB), blk, 0, stream>>>(WqF, xF, outq, 8, NSP,
                                                             2048L * 512, (long)CQ * NSP);
  wmma_gemm_frag<<<dim3(64, CQ / 16,  NB), blk, 0, stream>>>(WkF, xF, outk, 8, NSP,
                                                             2048L * 512, (long)CQ * NSP);
  wmma_gemm_frag<<<dim3(64, CCH / 16, NB), blk, 0, stream>>>(WvF, xF, outv, 8, NSP,
                                                             2048L * 512, (long)CCH * NSP);
  // stage attention operands
  stage_afrag<<<dim3(64, NB), blk, 0, stream>>>(outq, qF, 1, 1, NSP,
                                                (long)CQ * NSP, 256L * 512);   // Q as A
  stage_bfrag<<<dim3(64, NB), blk, 0, stream>>>(outk, kF, 1, NSP, 1,
                                                (long)CQ * NSP, 256L * 512);   // K as B
  stage_bfrag<<<dim3(512, NB), blk, 0, stream>>>(outv, vF, 128, 1, NSP,
                                                 (long)CCH * NSP, 2048L * 512); // V as B
  // flash attention -> pre[b][n][c]
  attn_flash<<<dim3(2, NSP / 16, NB), dim3(32), 0, stream>>>(qF, kF, vF, pre);
  // stage pre as B: B[k=c][col=n] = pre[n*C+c]
  stage_bfrag<<<dim3(512, NB), blk, 0, stream>>>(pre, preF, 8, 1, CCH,
                                                 (long)NSP * CCH, 2048L * 512);
  // y2 = Wp * pre
  wmma_gemm_frag<<<dim3(64, CCH / 16, NB), blk, 0, stream>>>(WpF, preF, y2, 8, NSP,
                                                             2048L * 512, (long)CCH * NSP);
  // instance norm + residual
  instnorm_add<<<dim3(CCH, NB), dim3(256), 0, stream>>>(y2, x, outa);
}
